// RCF_11965778886733
// MI455X (gfx1250) — matrix operand order repaired and checked by reading, not compiled, not run on hardware
//
#include <hip/hip_runtime.h>
#include <cmath>

#define EPSF 1e-5f

typedef _Float16 h8  __attribute__((ext_vector_type(8)));
typedef _Float16 h16 __attribute__((ext_vector_type(16)));
typedef float    f8  __attribute__((ext_vector_type(8)));
typedef unsigned int u32x4 __attribute__((ext_vector_type(4)));
typedef int          i32x8 __attribute__((ext_vector_type(8)));
typedef int          i32x4 __attribute__((ext_vector_type(4)));

__device__ __forceinline__ float sigmoidf_(float x) { return 1.0f / (1.0f + __expf(-x)); }

__device__ __forceinline__ float wave_sum32(float v) {
    // wave32 reduction (warpSize == 32 on gfx1250)
    for (int o = 16; o > 0; o >>= 1) v += __shfl_down(v, o, 32);
    return v;
}

// ---------------------------------------------------------------------------
// 0) zero outputs + scan u_idx into segment boundaries: idxs[0]=nseg, idxs[1..]=ends
// ---------------------------------------------------------------------------
__global__ void rcf_seg_init_kernel(const int* __restrict__ u_idx, int* __restrict__ idxs,
                                    float* __restrict__ out, int N) {
    if (blockIdx.x == 0 && threadIdx.x == 0) {
        out[0] = 0.0f;
        out[1] = 0.0f;
        int cnt = 0;
        int cur = u_idx[0];
        for (int i = 1; i < N; ++i) {
            int v = u_idx[i];
            if (v != cur) { idxs[1 + cnt] = i; ++cnt; cur = v; }
        }
        idxs[1 + cnt] = N; ++cnt;
        idxs[0] = cnt;
    }
}

// ---------------------------------------------------------------------------
// 1) gather embeddings; produce f32 rows, f16 rows (A operands), f16 u_e^T (B operand)
// ---------------------------------------------------------------------------
__global__ void rcf_prep_embed_kernel(const int* __restrict__ u_idx, const int* __restrict__ i_idx,
                                      const float* __restrict__ u_emb, const float* __restrict__ i_emb,
                                      const float* __restrict__ u_bias, const float* __restrict__ i_bias,
                                      float* __restrict__ u_e_f, float* __restrict__ i_e_f,
                                      _Float16* __restrict__ i_e_h, _Float16* __restrict__ u_eT_h,
                                      float* __restrict__ u_b_f, float* __restrict__ i_b_f,
                                      int N, int E) {
    int n = blockIdx.x;
    int u = u_idx[n];
    int it = i_idx[n];
    for (int e = threadIdx.x; e < E; e += blockDim.x) {
        float uv = u_emb[(size_t)u * E + e];
        float iv = i_emb[(size_t)it * E + e];
        u_e_f[(size_t)n * E + e] = uv;
        i_e_f[(size_t)n * E + e] = iv;
        i_e_h[(size_t)n * E + e] = (_Float16)iv;
        u_eT_h[(size_t)e * N + n] = (_Float16)uv;   // transposed layout [E][N]
    }
    if (threadIdx.x == 0) {
        u_b_f[n] = u_bias[u];
        i_b_f[n] = i_bias[it];
    }
}

// ---------------------------------------------------------------------------
// 2) transpose w_ih (3H x E) -> f16 [E][3H] for WMMA B operand
// ---------------------------------------------------------------------------
__global__ void rcf_prep_wih_kernel(const float* __restrict__ w_ih, _Float16* __restrict__ wihT,
                                    int E, int G) {
    int idx = blockIdx.x * blockDim.x + threadIdx.x;
    if (idx < E * G) {
        int e = idx / G;
        int j = idx % G;
        wihT[(size_t)e * G + j] = (_Float16)w_ih[(size_t)j * E + e];
    }
}

// ---------------------------------------------------------------------------
// WMMA A-fragment loader from a row-major f16 stripe (base = row 0 of 16-row tile)
// A per-lane layout (16x32 f16): row = l&15, kb = (l<16)?0:8, K = kb..kb+7, kb+16..kb+23
// ---------------------------------------------------------------------------
template <typename PTR>
__device__ __forceinline__ h16 rcf_load_a(PTR A, int lda, int row, int k, int kb) {
    h8 a0 = *(const h8*)(A + (size_t)row * lda + k + kb);
    h8 a1 = *(const h8*)(A + (size_t)row * lda + k + kb + 16);
    return __builtin_shufflevector(a0, a1, 0, 1, 2, 3, 4, 5, 6, 7,
                                           8, 9, 10, 11, 12, 13, 14, 15);
}

// ---------------------------------------------------------------------------
// TDM: DMA a [16 x E] f16 row-major stripe from global into LDS (2-D descriptor).
// D# layout per CDNA5 ISA ch.8 (group0: count/lds/global/type, group1: dims/strides).
// 6-arg builtin (clang-23 / therock HIP headers); groups 2/3 + trailing group zero
// (2-D tensor -> VADDR2/3 NULL-equivalent).
// ---------------------------------------------------------------------------
__device__ __forceinline__ void rcf_tdm_load_stripe16(unsigned lds_off,
                                                      const _Float16* gsrc, int E) {
    unsigned long long ga = (unsigned long long)(uintptr_t)gsrc;
    u32x4 g0;
    g0[0] = 1u;                                             // count=1, user descriptor
    g0[1] = lds_off;                                        // lds_addr (bytes)
    g0[2] = (unsigned)(ga & 0xFFFFFFFFu);                   // global_addr[31:0]
    g0[3] = (unsigned)((ga >> 32) & 0x01FFFFFFu) | (2u << 30); // global_addr[56:32] | type=2
    i32x8 g1;
    g1[0] = (int)(1u << 16);                                // data_size=1 (2 bytes)
    g1[1] = (int)(((unsigned)E & 0xFFFFu) << 16);           // tensor_dim0[15:0] @ [63:48]
    g1[2] = (int)((((unsigned)E >> 16) & 0xFFFFu) | (16u << 16)); // dim0 hi | tensor_dim1=16
    g1[3] = (int)(((unsigned)E & 0xFFFFu) << 16);           // tile_dim0=E @ [127:112]
    g1[4] = 16;                                             // tile_dim1=16 rows
    g1[5] = E;                                              // tensor_dim0_stride (low 32)
    g1[6] = 0;
    g1[7] = 0;
    i32x4 z4 = {0, 0, 0, 0};
    i32x8 z8 = {0, 0, 0, 0, 0, 0, 0, 0};
    __builtin_amdgcn_tensor_load_to_lds(g0, g1, z4, z4, z8, 0);
}

// ---------------------------------------------------------------------------
// 3) gi = i_e @ w_ih^T + b_ih  -> f32 [N][G]   (WMMA GEMM, one tile per wave)
// ---------------------------------------------------------------------------
__global__ void rcf_gi_gemm_kernel(const _Float16* __restrict__ i_e_h,
                                   const _Float16* __restrict__ wihT,
                                   const float* __restrict__ b_ih,
                                   float* __restrict__ gi, int N, int E, int G) {
    int lane = threadIdx.x & 31;
    int wave = (blockIdx.x * blockDim.x + threadIdx.x) >> 5;
    int wstride = (gridDim.x * blockDim.x) >> 5;
    int tn = N >> 4, tg = G >> 4;
    int total = tn * tg;
    const int kb = (lane < 16) ? 0 : 8;
    for (int tile = wave; tile < total; tile += wstride) {
        int bi = tile / tg, bj = tile % tg;
        int row = bi * 16 + (lane & 15);
        int col0 = bj * 16;
        f8 c = {};
        for (int k = 0; k < E; k += 32) {
            h16 a = rcf_load_a(i_e_h, E, row, k, kb);
            h16 b;
            for (int s = 0; s < 16; ++s) b[s] = wihT[(size_t)(k + kb + (s < 8 ? s : s + 8)) * G + col0 + (lane & 15)];
            c = __builtin_amdgcn_wmma_f32_16x16x32_f16(false, a, false, b, (short)0, c,
                                                       false, false);
        }
        int ncol = col0 + (lane & 15);
        float bias = b_ih[ncol];
        int m0 = bi * 16 + ((lane < 16) ? 0 : 8);
        #pragma unroll
        for (int v = 0; v < 8; ++v)
            gi[(size_t)(m0 + v) * G + ncol] = c[v] + bias;
    }
}

// ---------------------------------------------------------------------------
// 4) pos_rating_loss: sum log(eps + sigmoid(i_e @ u_e^T + u_b[:,None] + i_b[None,:]))
//    One workgroup per 16-row A stripe: stripe DMA'd to LDS once via TDM, 8 waves
//    sweep the 128 column tiles (WMMA, A from LDS, B from global), log-sigmoid
//    reduced in-register, atomic scalar accumulate.
// ---------------------------------------------------------------------------
__global__ void rcf_rating_gemm_kernel(const _Float16* __restrict__ i_e_h,
                                       const _Float16* __restrict__ u_eT_h,
                                       const float* __restrict__ u_b_f,
                                       const float* __restrict__ i_b_f,
                                       float* __restrict__ out, int N, int E) {
    __shared__ _Float16 sA[16 * 256];   // one 16 x E (E<=256) f16 stripe
    const int tid = threadIdx.x;
    const int lane = tid & 31;
    const int w = tid >> 5;
    const int nwaves = blockDim.x >> 5;
    const int tn = N >> 4;
    const int kb = (lane < 16) ? 0 : 8;
    const int arow = lane & 15;
    float acc = 0.0f;

    for (int bi = blockIdx.x; bi < tn; bi += gridDim.x) {
        if (w == 0) {   // one TDM DMA per stripe; TENSORcnt is per-wave
            rcf_tdm_load_stripe16((unsigned)(uintptr_t)sA,
                                  i_e_h + (size_t)bi * 16 * E, E);
            __builtin_amdgcn_s_wait_tensorcnt(0);
        }
        __syncthreads();   // publish LDS stripe to all waves

        int m0 = bi * 16 + ((lane < 16) ? 0 : 8);
        float ub0 = u_b_f[m0 + 0], ub1 = u_b_f[m0 + 1], ub2 = u_b_f[m0 + 2], ub3 = u_b_f[m0 + 3];
        float ub4 = u_b_f[m0 + 4], ub5 = u_b_f[m0 + 5], ub6 = u_b_f[m0 + 6], ub7 = u_b_f[m0 + 7];

        for (int bj = w; bj < tn; bj += nwaves) {
            int col0 = bj * 16;
            f8 c = {};
            for (int k = 0; k < E; k += 32) {
                h16 a = rcf_load_a((const _Float16*)sA, E, arow, k, kb);  // ds_load
                h16 b;
                for (int s = 0; s < 16; ++s) b[s] = u_eT_h[(size_t)(k + kb + (s < 8 ? s : s + 8)) * N + col0 + (lane & 15)];
                c = __builtin_amdgcn_wmma_f32_16x16x32_f16(false, a, false, b, (short)0, c,
                                                           false, false);
            }
            float ibv = i_b_f[col0 + (lane & 15)];
            acc += __logf(EPSF + sigmoidf_(c[0] + ub0 + ibv));
            acc += __logf(EPSF + sigmoidf_(c[1] + ub1 + ibv));
            acc += __logf(EPSF + sigmoidf_(c[2] + ub2 + ibv));
            acc += __logf(EPSF + sigmoidf_(c[3] + ub3 + ibv));
            acc += __logf(EPSF + sigmoidf_(c[4] + ub4 + ibv));
            acc += __logf(EPSF + sigmoidf_(c[5] + ub5 + ibv));
            acc += __logf(EPSF + sigmoidf_(c[6] + ub6 + ibv));
            acc += __logf(EPSF + sigmoidf_(c[7] + ub7 + ibv));
        }
        __syncthreads();   // stripe reads done before next bi overwrites sA
    }
    acc = wave_sum32(acc);
    if (lane == 0) atomicAdd(out + 0, acc);
}

// ---------------------------------------------------------------------------
// 5) neg_rating_loss: sum over (n,m) of log(eps + sigmoid(-(n_i_e . u_e + u_b + i_b))) / M
//    one wave per (n,m) pair, grid-stride
// ---------------------------------------------------------------------------
__global__ void rcf_neg_loss_kernel(const int* __restrict__ n_i_idx,
                                    const float* __restrict__ i_emb,
                                    const float* __restrict__ i_bias,
                                    const float* __restrict__ u_e_f,
                                    const float* __restrict__ u_b_f,
                                    float* __restrict__ out, int N, int M, int E) {
    int lane = threadIdx.x & 31;
    int wave = (blockIdx.x * blockDim.x + threadIdx.x) >> 5;
    int wstride = (gridDim.x * blockDim.x) >> 5;
    int total = N * M;
    float acc = 0.0f;
    for (int p = wave; p < total; p += wstride) {
        int n = p / M;
        int item = n_i_idx[p];
        const float* er = i_emb + (size_t)item * E;
        const float* ur = u_e_f + (size_t)n * E;
        if (p + wstride < total) {  // hint prefetch for next gather row
            __builtin_prefetch(i_emb + (size_t)n_i_idx[p + wstride] * E + lane, 0, 1);
        }
        float s = 0.0f;
        for (int e = lane; e < E; e += 32) s += er[e] * ur[e];
        s = wave_sum32(s);
        if (lane == 0) {
            float x = s + u_b_f[n] + i_bias[item];
            acc += __logf(EPSF + sigmoidf_(-x));   // 1 - sigmoid(x) == sigmoid(-x)
        }
    }
    if (lane == 0) atomicAdd(out + 0, acc / (float)M);
}

// ---------------------------------------------------------------------------
// 6) GRU segments (reference semantics incl. s0 = i-1), one workgroup per segment.
//    blockDim.x must equal E (== 256). h, hs staged in LDS.
// ---------------------------------------------------------------------------
__global__ void rcf_gru_seg_kernel(const float* __restrict__ gi,     // [N][G]
                                   const float* __restrict__ w_hh,   // [G][E]
                                   const float* __restrict__ b_hh,   // [G]
                                   const float* __restrict__ i_e_f,  // [N][E]
                                   const float* __restrict__ u_e_f,  // [N][E]
                                   const int* __restrict__ n_i_idx,  // [N][M]
                                   const float* __restrict__ i_emb,  // [NI][E]
                                   const int* __restrict__ idxs,
                                   float* __restrict__ out, int M, int E) {
    __shared__ float h[256];
    __shared__ float hs[256];
    __shared__ float xred[8];
    const int tid = threadIdx.x;
    const int lane = tid & 31;
    const int w = tid >> 5;
    const int nwaves = blockDim.x >> 5;
    const int G = 3 * E;
    const int nseg = idxs[0];

    float pos_acc = 0.0f;   // meaningful on tid 0
    float neg_acc = 0.0f;   // meaningful on lane 0 of each wave

    for (int s = blockIdx.x; s < nseg; s += gridDim.x) {
        const int s0 = (s == 0) ? 0 : (s - 1);
        const int end = idxs[1 + s];
        h[tid] = 0.0f;
        __syncthreads();
        for (int t = s0; t < end; ++t) {
            // gh = w_hh @ h + b_hh ; thread tid owns rows tid, tid+E, tid+2E
            float gh0 = b_hh[tid], gh1 = b_hh[tid + E], gh2 = b_hh[tid + 2 * E];
            const float* w0 = w_hh + (size_t)tid * E;
            const float* w1 = w_hh + (size_t)(tid + E) * E;
            const float* w2 = w_hh + (size_t)(tid + 2 * E) * E;
            #pragma unroll 4
            for (int k = 0; k < 256; ++k) {
                float hk = h[k];
                gh0 += w0[k] * hk;
                gh1 += w1[k] * hk;
                gh2 += w2[k] * hk;
            }
            // gates for dim d = tid
            const size_t gb = (size_t)t * G;
            float r  = sigmoidf_(gi[gb + tid] + gh0);
            float z  = sigmoidf_(gi[gb + E + tid] + gh1);
            float nn = tanhf(gi[gb + 2 * E + tid] + r * gh2);
            float hold = h[tid];
            float hnew = (1.0f - z) * nn + z * hold;
            __syncthreads();                    // all reads of old h done
            h[tid] = hnew;
            float hsv = hnew + u_e_f[(size_t)t * E + tid];
            hs[tid] = hsv;
            float xv = i_e_f[(size_t)t * E + tid];
            __syncthreads();                    // h, hs visible

            // pos: block-wide dot(hs, i_e[t])
            float p = wave_sum32(hsv * xv);
            if (lane == 0) xred[w] = p;
            __syncthreads();
            if (tid == 0) {
                float dot = 0.0f;
                for (int q = 0; q < nwaves; ++q) dot += xred[q];
                pos_acc += __logf(EPSF + sigmoidf_(dot));
            }

            // neg: wave w handles m = w, w+nwaves, ... ; n_mm = -(n_i_e . hs)
            for (int m = w; m < M; m += nwaves) {
                int item = n_i_idx[(size_t)t * M + m];
                const float* er = i_emb + (size_t)item * E;
                __builtin_prefetch(er + lane, 0, 1);
                float sdot = 0.0f;
                for (int e = lane; e < E; e += 32) sdot += er[e] * hs[e];
                sdot = wave_sum32(sdot);
                if (lane == 0) neg_acc += __logf(EPSF + sigmoidf_(-sdot));
            }
            __syncthreads();                    // hs reads done before next overwrite
        }
    }
    if (tid == 0) atomicAdd(out + 1, pos_acc);
    if (lane == 0) atomicAdd(out + 1, neg_acc / (float)M);
}

// ---------------------------------------------------------------------------
// launch
// ---------------------------------------------------------------------------
extern "C" void kernel_launch(void* const* d_in, const int* in_sizes, int n_in,
                              void* d_out, int out_size, void* d_ws, size_t ws_size,
                              hipStream_t stream) {
    const int*   u_idx   = (const int*)d_in[0];
    const int*   i_idx   = (const int*)d_in[1];
    const int*   n_i_idx = (const int*)d_in[2];
    const float* u_emb   = (const float*)d_in[4];
    const float* i_emb   = (const float*)d_in[5];
    const float* u_bias  = (const float*)d_in[6];
    const float* i_bias  = (const float*)d_in[7];
    const float* w_ih    = (const float*)d_in[8];
    const float* w_hh    = (const float*)d_in[9];
    const float* b_ih    = (const float*)d_in[10];
    const float* b_hh    = (const float*)d_in[11];
    float* out = (float*)d_out;

    const int N = in_sizes[1];
    const int M = in_sizes[2] / N;
    const int G = in_sizes[10];    // 3H
    const int E = G / 3;           // H == E

    // workspace carve-up (256B aligned blocks)
    char* base = (char*)d_ws;
    size_t off = 0;
    auto take = [&](size_t bytes) -> char* {
        char* p = base + off;
        off += (bytes + 255) & ~(size_t)255;
        return p;
    };
    float*    u_e_f  = (float*)take((size_t)N * E * 4);
    float*    i_e_f  = (float*)take((size_t)N * E * 4);
    _Float16* i_e_h  = (_Float16*)take((size_t)N * E * 2);
    _Float16* u_eT_h = (_Float16*)take((size_t)E * N * 2);
    _Float16* wihT   = (_Float16*)take((size_t)E * G * 2);
    float*    u_b_f  = (float*)take((size_t)N * 4);
    float*    i_b_f  = (float*)take((size_t)N * 4);
    float*    gi     = (float*)take((size_t)N * G * 4);
    int*      idxs   = (int*)take((size_t)(N + 2) * 4);
    (void)ws_size; (void)out_size; (void)n_in;

    // 0) segment scan + zero outputs
    rcf_seg_init_kernel<<<1, 32, 0, stream>>>(u_idx, idxs, out, N);
    // 1) gathers + f16 conversions + u_e transpose
    rcf_prep_embed_kernel<<<N, 256, 0, stream>>>(u_idx, i_idx, u_emb, i_emb, u_bias, i_bias,
                                                 u_e_f, i_e_f, i_e_h, u_eT_h, u_b_f, i_b_f, N, E);
    // 2) w_ih transpose to f16
    rcf_prep_wih_kernel<<<(E * G + 255) / 256, 256, 0, stream>>>(w_ih, wihT, E, G);
    // 3) gi GEMM (WMMA):  (N/16)*(G/16) tiles, 8 waves/block
    {
        int tiles = (N / 16) * (G / 16);
        int blocks = (tiles + 7) / 8;
        rcf_gi_gemm_kernel<<<blocks, 256, 0, stream>>>(i_e_h, wihT, b_ih, gi, N, E, G);
    }
    // 4) rating GEMM + log-sigmoid reduction (TDM stripe -> LDS, WMMA)
    {
        int blocks = N / 16;   // one workgroup per 16-row A stripe
        rcf_rating_gemm_kernel<<<blocks, 256, 0, stream>>>(i_e_h, u_eT_h, u_b_f, i_b_f, out, N, E);
    }
    // 5) negative-sample rating loss
    {
        int blocks = 1024;
        rcf_neg_loss_kernel<<<blocks, 256, 0, stream>>>(n_i_idx, i_emb, i_bias, u_e_f, u_b_f,
                                                        out, N, M, E);
    }
    // 6) GRU over segments (blockDim must equal E)
    rcf_gru_seg_kernel<<<32, E, 0, stream>>>(gi, w_hh, b_hh, i_e_f, u_e_f, n_i_idx, i_emb,
                                             idxs, out, M, E);
}
